// SwinTransformerBlock_10445360463957
// MI455X (gfx1250) — matrix operand order repaired and checked by reading, not compiled, not run on hardware
//
#include <hip/hip_runtime.h>
#include <hip/hip_bf16.h>
#include <math.h>

// Problem constants (static in the reference)
#define BB    8
#define LL    4096
#define CC    512
#define HH    8
#define WS    64
#define SS    32
#define HD    64            // CC/HH
#define MTOK  (BB*LL)       // 32768 tokens

typedef __attribute__((ext_vector_type(16))) __bf16   bf16x16;
typedef __attribute__((ext_vector_type(8)))  float    f32x8;
typedef __attribute__((ext_vector_type(4)))  unsigned u32x4;

union Frag { u32x4 u[2]; bf16x16 v; };

__device__ __forceinline__ unsigned short f2bf(float f) {
  unsigned u = __float_as_uint(f);
  unsigned r = u + 0x7FFFu + ((u >> 16) & 1u);   // round-to-nearest-even
  return (unsigned short)(r >> 16);
}

// A fragment (16x32 bf16, M x K), A row-major with leading dim ld.
// Lane layout (ISA 7.12.2): lanes 0-15 row m hold K = {0..7, 16..23},
// lanes 16-31 row m hold K = {8..15, 24..31}. Two 16B loads per lane.
__device__ __forceinline__ bf16x16 load_frag_a(const unsigned short* tile, int ld, int lane) {
  int m = lane & 15, hi = lane >> 4;
  const unsigned short* p = tile + (size_t)m * ld + hi * 8;
  Frag f;
  f.u[0] = *(const u32x4*)(p);        // K = k0 + 8*hi + 0..7
  f.u[1] = *(const u32x4*)(p + 16);   // K = k0 + 16 + 8*hi + 0..7
  return f.v;
}

// B fragment (32x16 bf16, K x N). We store B K-contiguous per column
// (weights pre-transposed to N x K; k/vT tiles are naturally K-contiguous),
// so column n per lane: lanes 0-15 K={0..7,16..23}, 16-31 K={8..15,24..31} (matches A).
__device__ __forceinline__ bf16x16 load_frag_b(const unsigned short* tile, int ld, int lane) {
  int n = lane & 15, hi = lane >> 4;
  const unsigned short* p = tile + (size_t)n * ld + hi * 8;
  Frag f;
  f.u[0] = *(const u32x4*)(p);
  f.u[1] = *(const u32x4*)(p + 16);
  return f.v;
}

#define WMMA_BF16(a, b, c) \
  __builtin_amdgcn_wmma_f32_16x16x32_bf16(false, (a), false, (b), (short)0, (c), false, false)

// ---------------- weight convert + transpose: (K,N) f32 -> (N,K) bf16 ----------
__global__ void k_wt(const float* __restrict__ w, unsigned short* __restrict__ wt,
                     int K, int N) {
  int idx = blockIdx.x * blockDim.x + threadIdx.x;
  if (idx >= K * N) return;
  int k = idx / N, n = idx - k * N;
  wt[(size_t)n * K + k] = f2bf(w[idx]);
}

// ---------------- LayerNorm (+optional roll by `roll`) -> bf16 rows ------------
// Output row r holds LN(x[batch][(l + roll) mod L]) (roll=SS implements roll(h,-SS)).
__global__ void k_ln(const float* __restrict__ x, const float* __restrict__ g,
                     const float* __restrict__ b, unsigned short* __restrict__ out,
                     int roll) {
  int wid  = (blockIdx.x * blockDim.x + threadIdx.x) >> 5;
  int lane = threadIdx.x & 31;
  if (wid >= MTOK) return;
  int batch = wid >> 12;                 // / LL
  int lo    = wid & (LL - 1);
  int li    = (lo + roll) & (LL - 1);
  const float* src = x + ((size_t)batch * LL + li) * CC;
  float vals[16];
  float s = 0.f, sq = 0.f;
#pragma unroll
  for (int t = 0; t < 16; ++t) {
    float v = src[lane + 32 * t];
    vals[t] = v; s += v; sq += v * v;
  }
#pragma unroll
  for (int off = 16; off >= 1; off >>= 1) {
    s  += __shfl_xor(s,  off, 32);
    sq += __shfl_xor(sq, off, 32);
  }
  float mean = s * (1.f / CC);
  float var  = sq * (1.f / CC) - mean * mean;
  float rstd = rsqrtf(var + 1e-5f);
  unsigned short* dst = out + (size_t)wid * CC;
#pragma unroll
  for (int t = 0; t < 16; ++t) {
    int c = lane + 32 * t;
    dst[c] = f2bf((vals[t] - mean) * rstd * g[c] + b[c]);
  }
}

// ---------------- WMMA GEMM: C[M,N] = A[M,K](bf16) * Bt[N,K](bf16) + bias -------
// One wave computes a 64x64 tile (4x4 WMMA tiles, K step 32).
enum { EPI_BF16 = 0, EPI_PROJ = 1, EPI_GELU = 2, EPI_FC2 = 3 };

template <int EPI>
__global__ void __launch_bounds__(128)
k_gemm(const unsigned short* __restrict__ A, const unsigned short* __restrict__ Bt,
       const float* __restrict__ bias, unsigned short* __restrict__ outb,
       float* __restrict__ outf, const float* __restrict__ res,
       int M, int N, int K) {
  int wid  = (blockIdx.x * blockDim.x + threadIdx.x) >> 5;
  int lane = threadIdx.x & 31;
  int tilesN = N >> 6;
  int totalW = (M >> 6) * tilesN;
  if (wid >= totalW) return;
  int tM = (wid / tilesN) << 6;
  int tN = (wid % tilesN) << 6;

  f32x8 zero = {0.f, 0.f, 0.f, 0.f, 0.f, 0.f, 0.f, 0.f};
  f32x8 acc[4][4];
#pragma unroll
  for (int i = 0; i < 4; ++i)
#pragma unroll
    for (int j = 0; j < 4; ++j) acc[i][j] = zero;

  for (int k0 = 0; k0 < K; k0 += 32) {
    bf16x16 fa[4], fb[4];
#pragma unroll
    for (int i = 0; i < 4; ++i)
      fa[i] = load_frag_a(A + (size_t)(tM + 16 * i) * K + k0, K, lane);
#pragma unroll
    for (int j = 0; j < 4; ++j)
      fb[j] = load_frag_b(Bt + (size_t)(tN + 16 * j) * K + k0, K, lane);
#pragma unroll
    for (int i = 0; i < 4; ++i)
#pragma unroll
      for (int j = 0; j < 4; ++j)
        acc[i][j] = WMMA_BF16(fa[i], fb[j], acc[i][j]);
  }

  int lhi = lane >> 4, lno = lane & 15;
#pragma unroll
  for (int i = 0; i < 4; ++i)
#pragma unroll
    for (int j = 0; j < 4; ++j)
#pragma unroll
      for (int r = 0; r < 8; ++r) {
        int m = tM + 16 * i + r + 8 * lhi;     // C/D layout: M = vgpr + 8*(lane>>4)
        int n = tN + 16 * j + lno;             //             N = lane & 15
        float v = acc[i][j][r] + bias[n];
        if (EPI == EPI_BF16) {
          outb[(size_t)m * N + n] = f2bf(v);
        } else if (EPI == EPI_PROJ) {
          // reverse roll: rolled row m -> original l = (m%L + SS) mod L; add shortcut x
          int batch = m >> 12;
          int l = ((m & (LL - 1)) + SS) & (LL - 1);
          size_t dst = ((size_t)batch * LL + l) * CC + n;
          outf[dst] = res[dst] + v;
        } else if (EPI == EPI_GELU) {
          float gv = 0.5f * v * (1.f + erff(v * 0.70710678118654752f));
          outb[(size_t)m * N + n] = f2bf(gv);
        } else { // EPI_FC2: + residual x2 -> final output f32
          outf[(size_t)m * N + n] = res[(size_t)m * N + n] + v;
        }
      }
}

// ---------------- attention: one wave per (window, head) ------------------------
__global__ void __launch_bounds__(32)
k_attn(const unsigned short* __restrict__ qkv, const float* __restrict__ rpb,
       unsigned short* __restrict__ attn_out) {
  __shared__ __attribute__((aligned(16))) float          Ssm[WS * WS];   // 16KB
  __shared__ __attribute__((aligned(16))) unsigned short Psm[WS * WS];   // 8KB
  __shared__ __attribute__((aligned(16))) unsigned short VTs[WS * WS];   // 8KB

  int wh   = blockIdx.x;        // 0..4095 = 512 windows * 8 heads
  int bwin = wh >> 3;
  int h    = wh & 7;
  int lane = threadIdx.x & 31;
  int base = bwin * WS;         // first (rolled) token row of this window

  const unsigned short* qb = qkv + (size_t)base * (3 * CC) + h * HD;
  const unsigned short* kb = qb + CC;
  const unsigned short* vb = qb + 2 * CC;

  f32x8 zero = {0.f, 0.f, 0.f, 0.f, 0.f, 0.f, 0.f, 0.f};
  f32x8 acc[4][4];
#pragma unroll
  for (int i = 0; i < 4; ++i)
#pragma unroll
    for (int j = 0; j < 4; ++j) acc[i][j] = zero;

  // S = q @ k^T  (k row-major IS k^T in the B-fragment layout: column n = k row n)
  for (int k0 = 0; k0 < HD; k0 += 32) {
    bf16x16 fa[4], fb[4];
#pragma unroll
    for (int i = 0; i < 4; ++i)
      fa[i] = load_frag_a(qb + (size_t)(16 * i) * (3 * CC) + k0, 3 * CC, lane);
#pragma unroll
    for (int j = 0; j < 4; ++j)
      fb[j] = load_frag_b(kb + (size_t)(16 * j) * (3 * CC) + k0, 3 * CC, lane);
#pragma unroll
    for (int i = 0; i < 4; ++i)
#pragma unroll
      for (int j = 0; j < 4; ++j)
        acc[i][j] = WMMA_BF16(fa[i], fb[j], acc[i][j]);
  }

  int lhi = lane >> 4, lno = lane & 15;
#pragma unroll
  for (int i = 0; i < 4; ++i)
#pragma unroll
    for (int j = 0; j < 4; ++j)
#pragma unroll
      for (int r = 0; r < 8; ++r)
        Ssm[(16 * i + r + 8 * lhi) * WS + 16 * j + lno] = acc[i][j][r];

  // stage transposed V tile while S settles: VT[d][m] = v[m][d]
  for (int idx = lane; idx < WS * WS; idx += 32) {
    int m = idx >> 6, d = idx & 63;
    VTs[d * WS + m] = vb[(size_t)m * (3 * CC) + d];
  }
  asm volatile("" ::: "memory");

  // softmax (each lane owns rows lane and lane+32 exclusively -> no races)
  const float scale = 0.125f;                      // hd^-0.5
  bool masked = (bwin & 63) == 63;                 // shift mask hits only last window
#pragma unroll
  for (int rr = 0; rr < 2; ++rr) {
    int m = lane + 32 * rr;
    float vmax = -3.0e38f;
    for (int n = 0; n < WS; ++n) {
      float t = Ssm[m * WS + n] * scale + rpb[(m - n + WS - 1) * HH + h];
      if (masked && ((m < 32) != (n < 32))) t -= 100.f;
      Ssm[m * WS + n] = t;
      vmax = fmaxf(vmax, t);
    }
    float sum = 0.f;
    for (int n = 0; n < WS; ++n) {
      float e = __expf(Ssm[m * WS + n] - vmax);
      Ssm[m * WS + n] = e;
      sum += e;
    }
    float inv = 1.f / sum;
    for (int n = 0; n < WS; ++n)
      Psm[m * WS + n] = f2bf(Ssm[m * WS + n] * inv);
  }
  asm volatile("" ::: "memory");

  // O = P @ V  (A from LDS P row-major, B from LDS VT: K-contiguous rows)
#pragma unroll
  for (int i = 0; i < 4; ++i)
#pragma unroll
    for (int j = 0; j < 4; ++j) acc[i][j] = zero;
  for (int k0 = 0; k0 < WS; k0 += 32) {
    bf16x16 fa[4], fb[4];
#pragma unroll
    for (int i = 0; i < 4; ++i)
      fa[i] = load_frag_a(Psm + (16 * i) * WS + k0, WS, lane);
#pragma unroll
    for (int j = 0; j < 4; ++j)
      fb[j] = load_frag_b(VTs + (16 * j) * WS + k0, WS, lane);
#pragma unroll
    for (int i = 0; i < 4; ++i)
#pragma unroll
      for (int j = 0; j < 4; ++j)
        acc[i][j] = WMMA_BF16(fa[i], fb[j], acc[i][j]);
  }

#pragma unroll
  for (int i = 0; i < 4; ++i)
#pragma unroll
    for (int j = 0; j < 4; ++j)
#pragma unroll
      for (int r = 0; r < 8; ++r) {
        int m = 16 * i + r + 8 * lhi;
        int n = 16 * j + lno;
        attn_out[(size_t)(base + m) * CC + h * HD + n] = f2bf(acc[i][j][r]);
      }
}

// -------------------------------------------------------------------------------
extern "C" void kernel_launch(void* const* d_in, const int* in_sizes, int n_in,
                              void* d_out, int out_size, void* d_ws, size_t ws_size,
                              hipStream_t stream) {
  const float* x      = (const float*)d_in[0];
  const float* n1w    = (const float*)d_in[1];
  const float* n1b    = (const float*)d_in[2];
  const float* qkv_w  = (const float*)d_in[3];
  const float* qkv_b  = (const float*)d_in[4];
  const float* rpb    = (const float*)d_in[5];
  const float* proj_w = (const float*)d_in[6];
  const float* proj_b = (const float*)d_in[7];
  const float* n2w    = (const float*)d_in[8];
  const float* n2b    = (const float*)d_in[9];
  const float* fc1_w  = (const float*)d_in[10];
  const float* fc1_b  = (const float*)d_in[11];
  const float* fc2_w  = (const float*)d_in[12];
  const float* fc2_b  = (const float*)d_in[13];

  char* ws = (char*)d_ws;
  size_t off = 0;
  auto alloc = [&](size_t bytes) -> void* {
    void* p = ws + off;
    off = (off + bytes + 255) & ~(size_t)255;
    return p;
  };

  unsigned short* wt_qkv = (unsigned short*)alloc((size_t)CC * 3 * CC * 2);
  unsigned short* wt_prj = (unsigned short*)alloc((size_t)CC * CC * 2);
  unsigned short* wt_fc1 = (unsigned short*)alloc((size_t)CC * 4 * CC * 2);
  unsigned short* wt_fc2 = (unsigned short*)alloc((size_t)4 * CC * CC * 2);
  unsigned short* h1     = (unsigned short*)alloc((size_t)MTOK * CC * 2);
  unsigned short* qkv    = (unsigned short*)alloc((size_t)MTOK * 3 * CC * 2);
  unsigned short* att_o  = (unsigned short*)alloc((size_t)MTOK * CC * 2);
  float*          x2     = (float*)         alloc((size_t)MTOK * CC * 4);
  unsigned short* h2     = (unsigned short*)alloc((size_t)MTOK * CC * 2);
  unsigned short* mlp    = (unsigned short*)alloc((size_t)MTOK * 4 * CC * 2);
  (void)ws_size; (void)in_sizes; (void)n_in; (void)out_size;

  // weights -> bf16, transposed to (N, K)
  k_wt<<<(CC * 3 * CC + 255) / 256, 256, 0, stream>>>(qkv_w, wt_qkv, CC, 3 * CC);
  k_wt<<<(CC * CC + 255) / 256, 256, 0, stream>>>(proj_w, wt_prj, CC, CC);
  k_wt<<<(CC * 4 * CC + 255) / 256, 256, 0, stream>>>(fc1_w, wt_fc1, CC, 4 * CC);
  k_wt<<<(4 * CC * CC + 255) / 256, 256, 0, stream>>>(fc2_w, wt_fc2, 4 * CC, CC);

  // LN1 + roll(-SS) -> h1 (bf16, rolled token order)
  k_ln<<<MTOK / 8, 256, 0, stream>>>(x, n1w, n1b, h1, SS);

  // QKV GEMM: (32768 x 512) x (512 x 1536)
  {
    int waves = (MTOK / 64) * ((3 * CC) / 64);
    k_gemm<EPI_BF16><<<waves / 4, 128, 0, stream>>>(h1, wt_qkv, qkv_b, qkv,
                                                    nullptr, nullptr,
                                                    MTOK, 3 * CC, CC);
  }

  // windowed attention: 512 windows * 8 heads, one wave each
  k_attn<<<(MTOK / WS) * HH, 32, 0, stream>>>(qkv, rpb, att_o);

  // proj GEMM + reverse roll + shortcut: x2 = x + roll(out, +SS)
  {
    int waves = (MTOK / 64) * (CC / 64);
    k_gemm<EPI_PROJ><<<waves / 4, 128, 0, stream>>>(att_o, wt_prj, proj_b,
                                                    nullptr, x2, x,
                                                    MTOK, CC, CC);
  }

  // LN2 -> h2 (bf16)
  k_ln<<<MTOK / 8, 256, 0, stream>>>(x2, n2w, n2b, h2, 0);

  // FC1 + exact GELU
  {
    int waves = (MTOK / 64) * ((4 * CC) / 64);
    k_gemm<EPI_GELU><<<waves / 4, 128, 0, stream>>>(h2, wt_fc1, fc1_b, mlp,
                                                    nullptr, nullptr,
                                                    MTOK, 4 * CC, CC);
  }

  // FC2 + residual -> d_out (f32)
  {
    int waves = (MTOK / 64) * (CC / 64);
    k_gemm<EPI_FC2><<<waves / 4, 128, 0, stream>>>(mlp, wt_fc2, fc2_b,
                                                   nullptr, (float*)d_out, x2,
                                                   MTOK, CC, 4 * CC);
  }
}
